// Decoder_19507741458529
// MI455X (gfx1250) — compile-verified, matrix operand order checked by reading, not run on hardware
//
#include <hip/hip_runtime.h>
#include <math.h>

#define V_SZ   50000
#define E_SZ   256
#define ENC2_SZ 512
#define H_SZ   512
#define B_SZ   128
#define S_SZ   512

typedef __attribute__((ext_vector_type(16))) __bf16 v16bf;
typedef __attribute__((ext_vector_type(8)))  float  v8f;
typedef __attribute__((ext_vector_type(8)))  unsigned u32x8;

// ---------------------------------------------------------------- helpers

// two fp32 -> packed bf16x2 dword (truncation) in ONE v_perm_b32
__device__ __forceinline__ unsigned pack2(float lo, float hi) {
  return __builtin_amdgcn_perm(__builtin_bit_cast(unsigned, hi),
                               __builtin_bit_cast(unsigned, lo), 0x07060302u);
}

// K offset inside a 16x32 bf16 fragment for vgpr-pair i (ISA 7.12.2)
__device__ __forceinline__ int frag_k(int i, int half) {
  return 2 * (i & 3) + 16 * (i >> 2) + half * 8;
}

// A fragment: 16x32 tile of a row-major fp32 matrix -> bf16 (8 v_perm_b32)
__device__ __forceinline__ v16bf load_a_frag(const float* __restrict__ A, long lda,
                                             int m0, int k0) {
  int lane = threadIdx.x & 31;
  int m = lane & 15, half = lane >> 4;
  const float* row = A + (long)(m0 + m) * lda + k0 + half * 8;
  u32x8 a;
#pragma unroll
  for (int i = 0; i < 4; ++i) {
    a[i]     = pack2(row[2 * i],      row[2 * i + 1]);
    a[i + 4] = pack2(row[16 + 2 * i], row[16 + 2 * i + 1]);
  }
  return __builtin_bit_cast(v16bf, a);
}

// B fragment: 32x16 tile of a row-major (K x N) fp32 matrix
__device__ __forceinline__ v16bf load_b_frag_rm(const float* __restrict__ Bm, long ldb,
                                                int k0, int n0) {
  int lane = threadIdx.x & 31;
  int n = lane & 15, half = lane >> 4;
  const float* col = Bm + (long)(k0 + half * 8) * ldb + n0 + n;
  u32x8 b;
#pragma unroll
  for (int i = 0; i < 4; ++i) {
    b[i]     = pack2(col[(long)(2 * i) * ldb],      col[(long)(2 * i + 1) * ldb]);
    b[i + 4] = pack2(col[(long)(16 + 2 * i) * ldb], col[(long)(16 + 2 * i + 1) * ldb]);
  }
  return __builtin_bit_cast(v16bf, b);
}

// B fragment from W^T (W row-major N x K): identical per-lane layout to A-frag
__device__ __forceinline__ v16bf load_b_frag_wt(const float* __restrict__ W, long ldw,
                                                int k0, int n0) {
  return load_a_frag(W, ldw, n0, k0);
}

__device__ __forceinline__ v8f wmma_bf16(v16bf a, v16bf b, v8f c) {
  return __builtin_amdgcn_wmma_f32_16x16x32_bf16(false, a, false, b, (short)0, c,
                                                 false, false);
}

__device__ __forceinline__ float blk_max(float v, float* red, int nw) {
  int lane = threadIdx.x & 31, w = threadIdx.x >> 5;
#pragma unroll
  for (int m = 16; m >= 1; m >>= 1) v = fmaxf(v, __shfl_xor(v, m, 32));
  if (lane == 0) red[w] = v;
  __syncthreads();
  float t = red[0];
  for (int i = 1; i < nw; ++i) t = fmaxf(t, red[i]);
  __syncthreads();
  return t;
}

__device__ __forceinline__ float blk_sum(float v, float* red, int nw) {
  int lane = threadIdx.x & 31, w = threadIdx.x >> 5;
#pragma unroll
  for (int m = 16; m >= 1; m >>= 1) v += __shfl_xor(v, m, 32);
  if (lane == 0) red[w] = v;
  __syncthreads();
  float t = 0.f;
  for (int i = 0; i < nw; ++i) t += red[i];
  __syncthreads();
  return t;
}

// ---------------------------------------------------------------- kernels

// 0a) pre-swizzle Wa_e (Wa rows [512,1024)) into per-lane B-fragment layout:
//     dword d = ((nt*16 + ks)*32 + lane)*8 + i  ->  packed {k,k+1} column n
__global__ void __launch_bounds__(256) pack_wae_kernel(
    const float* __restrict__ Wa, unsigned* __restrict__ bbuf) {
  int d = blockIdx.x * 256 + threadIdx.x;          // 131072 dwords
  int i = d & 7, lane = (d >> 3) & 31, ks = (d >> 8) & 15, nt = d >> 12;
  int n = lane & 15, half = lane >> 4;
  int k = ks * 32 + frag_k(i, half);
  const float* p = Wa + (long)(512 + k) * 512 + nt * 16 + n;
  bbuf[d] = pack2(p[0], p[512]);
}

// 0b) pre-swizzle concat_out into per-lane A-fragment layout
__global__ void __launch_bounds__(256) pack_cat_kernel(
    const float* __restrict__ cat, unsigned* __restrict__ abuf) {
  int d = blockIdx.x * 256 + threadIdx.x;          // 32768 dwords
  int i = d & 7, lane = (d >> 3) & 31, ks = (d >> 8) & 15, mt = d >> 12;
  int m = mt * 16 + (lane & 15), half = lane >> 4;
  int k = ks * 32 + frag_k(i, half);
  const float* p = cat + (long)m * H_SZ + k;
  abuf[d] = pack2(p[0], p[1]);
}

// 1) GRU cell: one wave per 16x16 tile of (B=128, H=512)
__global__ void __launch_bounds__(256) gru_kernel(
    const float* __restrict__ x, const float* __restrict__ h,
    const float* __restrict__ Wih, const float* __restrict__ Whh,
    const float* __restrict__ bih, const float* __restrict__ bhh,
    float* __restrict__ hnew, float* __restrict__ hnew_out) {
  int wid = blockIdx.x * (blockDim.x >> 5) + (threadIdx.x >> 5);
  int mt = wid >> 5;   // 0..7
  int jt = wid & 31;   // 0..31
  int lane = threadIdx.x & 31;
  int n = lane & 15, half = lane >> 4;
  int col = jt * 16 + n;

  v8f gir, giz, gin, ghr, ghz, ghn;
#pragma unroll
  for (int r = 0; r < 8; ++r) {
    gir[r] = bih[col];        giz[r] = bih[col + 512];  gin[r] = bih[col + 1024];
    ghr[r] = bhh[col];        ghz[r] = bhh[col + 512];  ghn[r] = bhh[col + 1024];
  }
  for (int k = 0; k < E_SZ; k += 32) {
    v16bf a = load_a_frag(x, E_SZ, mt * 16, k);
    gir = wmma_bf16(a, load_b_frag_wt(Wih, E_SZ, k, jt * 16), gir);
    giz = wmma_bf16(a, load_b_frag_wt(Wih, E_SZ, k, 512 + jt * 16), giz);
    gin = wmma_bf16(a, load_b_frag_wt(Wih, E_SZ, k, 1024 + jt * 16), gin);
  }
  for (int k = 0; k < H_SZ; k += 32) {
    v16bf a = load_a_frag(h, H_SZ, mt * 16, k);
    ghr = wmma_bf16(a, load_b_frag_wt(Whh, H_SZ, k, jt * 16), ghr);
    ghz = wmma_bf16(a, load_b_frag_wt(Whh, H_SZ, k, 512 + jt * 16), ghz);
    ghn = wmma_bf16(a, load_b_frag_wt(Whh, H_SZ, k, 1024 + jt * 16), ghn);
  }
#pragma unroll
  for (int r = 0; r < 8; ++r) {
    int m = mt * 16 + r + 8 * half;
    float rg = 1.f / (1.f + expf(-(gir[r] + ghr[r])));
    float zg = 1.f / (1.f + expf(-(giz[r] + ghz[r])));
    float ng = tanhf(gin[r] + rg * ghn[r]);
    float hv = h[m * H_SZ + col];
    float o  = (1.f - zg) * ng + zg * hv;
    hnew[m * H_SZ + col]     = o;
    hnew_out[m * H_SZ + col] = o;
  }
}

// 2) hproj = h_new @ Wa_h + ba   (Wa rows [0,512))
__global__ void __launch_bounds__(256) hproj_kernel(
    const float* __restrict__ hnew, const float* __restrict__ Wa,
    const float* __restrict__ ba, float* __restrict__ hproj) {
  int wid = blockIdx.x * (blockDim.x >> 5) + (threadIdx.x >> 5);
  int mt = wid >> 5, jt = wid & 31;
  int lane = threadIdx.x & 31;
  int n = lane & 15, half = lane >> 4;
  v8f acc;
#pragma unroll
  for (int r = 0; r < 8; ++r) acc[r] = ba[jt * 16 + n];
  for (int k = 0; k < H_SZ; k += 32)
    acc = wmma_bf16(load_a_frag(hnew, H_SZ, mt * 16, k),
                    load_b_frag_rm(Wa, H_SZ, k, jt * 16), acc);
#pragma unroll
  for (int r = 0; r < 8; ++r)
    hproj[(mt * 16 + r + 8 * half) * H_SZ + jt * 16 + n] = acc[r];
}

// 3) att_energy[b,s] = tanh(enc[b,s,:] @ Wa_e + hproj[b,:]) . v
//    A-panel (16 rows x K=512) cached in registers -> text read once from HBM.
//    Pre-swizzled Wa_e panels staged to LDS with async-to-LDS b128 copies;
//    B-fragment = 32 contiguous bytes in LDS (two ds_load_b128).
__global__ void __launch_bounds__(256) att_kernel(
    const float* __restrict__ text, const unsigned* __restrict__ bbuf,
    const float* __restrict__ hproj, const float* __restrict__ vvec,
    float* __restrict__ attE) {
  __shared__ __align__(16) unsigned panel[4096];   // 16KB: one nt panel
  int wwave = threadIdx.x >> 5;
  int wid = blockIdx.x * 8 + wwave;
  int rowbase = wid * 16;
  int b  = rowbase >> 9;
  int s0 = rowbase & 511;
  int lane = threadIdx.x & 31;
  int n = lane & 15, half = lane >> 4;

  // enc[b, s0+m, e] = text[((s0+m)*B + b)*ENC2 + e]
  const float* Abase = text + (long)s0 * (B_SZ * ENC2_SZ) + (long)b * ENC2_SZ;
  v16bf afrag[16];
#pragma unroll
  for (int ks = 0; ks < 16; ++ks)
    afrag[ks] = load_a_frag(Abase, (long)B_SZ * ENC2_SZ, 0, ks * 32);

  float rowAcc[8];
#pragma unroll
  for (int r = 0; r < 8; ++r) rowAcc[r] = 0.f;

  for (int nt = 0; nt < 32; ++nt) {
    __syncthreads();
    {
      const unsigned* src = bbuf + nt * 4096;
#pragma unroll
      for (int it = 0; it < 4; ++it) {
        unsigned lds_addr = (unsigned)(unsigned long long)
                            (panel + threadIdx.x * 4 + it * 1024);
        const unsigned* g = src + threadIdx.x * 4 + it * 1024;
        asm volatile("global_load_async_to_lds_b128 %0, %1, off"
                     :: "v"(lds_addr), "v"(g) : "memory");
      }
      asm volatile("s_wait_asynccnt 0" ::: "memory");
    }
    __syncthreads();

    float hp = hproj[b * H_SZ + nt * 16 + n];
    v8f acc;
#pragma unroll
    for (int r = 0; r < 8; ++r) acc[r] = hp;
#pragma unroll
    for (int ks = 0; ks < 16; ++ks) {
      v16bf bf = *(const v16bf*)(panel + (ks * 32 + lane) * 8);
      acc = wmma_bf16(afrag[ks], bf, acc);
    }
    float vv = vvec[nt * 16 + n];
#pragma unroll
    for (int r = 0; r < 8; ++r) rowAcc[r] += tanhf(acc[r]) * vv;
  }
#pragma unroll
  for (int r = 0; r < 8; ++r) {
    float t = rowAcc[r];
    t += __shfl_xor(t, 1, 32);
    t += __shfl_xor(t, 2, 32);
    t += __shfl_xor(t, 4, 32);
    t += __shfl_xor(t, 8, 32);
    rowAcc[r] = t;
  }
  if (n == 0) {
#pragma unroll
    for (int r = 0; r < 8; ++r)
      attE[b * S_SZ + s0 + r + 8 * half] = rowAcc[r];
  }
}

// 4) softmax over S + weighted context (one block per batch element)
__global__ void __launch_bounds__(256) ctx_kernel(
    const float* __restrict__ attE, const float* __restrict__ text,
    float* __restrict__ wc) {
  __shared__ float sw[512];
  __shared__ float red[8];
  int b = blockIdx.x, tid = threadIdx.x;
  float e0 = attE[b * S_SZ + tid];
  float e1 = attE[b * S_SZ + tid + 256];
  float m = blk_max(fmaxf(e0, e1), red, 8);
  float x0 = expf(e0 - m), x1 = expf(e1 - m);
  float s = blk_sum(x0 + x1, red, 8);
  sw[tid]       = x0 / s;
  sw[tid + 256] = x1 / s;
  __syncthreads();
  float acc0 = 0.f, acc1 = 0.f;
  for (int si = 0; si < S_SZ; ++si) {
    float w = sw[si];
    const float* row = text + ((long)si * B_SZ + b) * ENC2_SZ;
    acc0 += w * row[tid];
    acc1 += w * row[tid + 256];
  }
  wc[b * ENC2_SZ + tid]       = acc0;
  wc[b * ENC2_SZ + tid + 256] = acc1;
}

// 5) p_gen = sigmoid([wc, h_new, x] @ Wp + bp), one wave per batch row
__global__ void __launch_bounds__(32) pgen_kernel(
    const float* __restrict__ wc, const float* __restrict__ hnew,
    const float* __restrict__ x, const float* __restrict__ Wp,
    const float* __restrict__ bp, float* __restrict__ pgen) {
  int b = blockIdx.x, lane = threadIdx.x;
  float acc = 0.f;
  for (int k = lane; k < 512; k += 32) acc += wc[b * 512 + k] * Wp[k];
  for (int k = lane; k < 512; k += 32) acc += hnew[b * 512 + k] * Wp[512 + k];
  for (int k = lane; k < 256; k += 32) acc += x[b * 256 + k] * Wp[1024 + k];
#pragma unroll
  for (int m = 16; m >= 1; m >>= 1) acc += __shfl_xor(acc, m, 32);
  if (lane == 0) pgen[b] = 1.f / (1.f + expf(-(acc + bp[0])));
}

// 6) concat_out = tanh([wc, h_new] @ Wc + bc)
__global__ void __launch_bounds__(256) concat_kernel(
    const float* __restrict__ wc, const float* __restrict__ hnew,
    const float* __restrict__ Wc, const float* __restrict__ bc,
    float* __restrict__ cat) {
  int wid = blockIdx.x * (blockDim.x >> 5) + (threadIdx.x >> 5);
  int mt = wid >> 5, jt = wid & 31;
  int lane = threadIdx.x & 31;
  int n = lane & 15, half = lane >> 4;
  v8f acc;
#pragma unroll
  for (int r = 0; r < 8; ++r) acc[r] = bc[jt * 16 + n];
  for (int k = 0; k < 1024; k += 32) {
    const float* A = (k < 512) ? wc : hnew;
    int ka = (k < 512) ? k : k - 512;
    acc = wmma_bf16(load_a_frag(A, 512, mt * 16, ka),
                    load_b_frag_rm(Wc, H_SZ, k, jt * 16), acc);
  }
#pragma unroll
  for (int r = 0; r < 8; ++r)
    cat[(mt * 16 + r + 8 * half) * H_SZ + jt * 16 + n] = tanhf(acc[r]);
}

// 7) pred = p_gen * (cat @ Wo + bo); zero copy-max buffer.
//    One wave per 16-wide vocab tile covering all 8 M tiles: Wo streams from
//    HBM once; A-fragments come pre-swizzled (two b128 loads from L2).
__global__ void __launch_bounds__(256) pred_kernel(
    const unsigned* __restrict__ abuf, const float* __restrict__ Wo,
    const float* __restrict__ bo, const float* __restrict__ pgen,
    float* __restrict__ pred, float* __restrict__ cpy) {
  int wid = blockIdx.x * 8 + (threadIdx.x >> 5);
  if (wid >= V_SZ / 16) return;
  int n0 = wid * 16;
  int lane = threadIdx.x & 31;
  int n = lane & 15, half = lane >> 4;
  v8f acc[8];
  float bias = bo[n0 + n];
#pragma unroll
  for (int mt = 0; mt < 8; ++mt)
#pragma unroll
    for (int r = 0; r < 8; ++r) acc[mt][r] = bias;
  for (int ks = 0; ks < 16; ++ks) {
    v16bf bf = load_b_frag_rm(Wo, V_SZ, ks * 32, n0);
#pragma unroll
    for (int mt = 0; mt < 8; ++mt) {
      v16bf af = *(const v16bf*)(abuf + ((mt * 16 + ks) * 32 + lane) * 8);
      acc[mt] = wmma_bf16(af, bf, acc[mt]);
    }
  }
#pragma unroll
  for (int mt = 0; mt < 8; ++mt) {
#pragma unroll
    for (int r = 0; r < 8; ++r) {
      int m = mt * 16 + r + 8 * half;
      long off = (long)m * V_SZ + n0 + n;
      pred[off] = pgen[m] * acc[mt][r];
      cpy[off]  = 0.f;
    }
  }
}

// 8) scatter-max of copy scores into vocab (int atomicMax on floats: base 0,
//    non-negative winners compare correctly as ints, negatives lose to 0)
__global__ void __launch_bounds__(256) scatter_kernel(
    const float* __restrict__ pgen, const float* __restrict__ attE,
    const int* __restrict__ ids, float* __restrict__ cpy) {
  int idx = blockIdx.x * blockDim.x + threadIdx.x;
  if (idx >= B_SZ * S_SZ) return;
  int b = idx >> 9, s = idx & 511;
  float val = (1.f - pgen[b]) * attE[b * S_SZ + s];
  int id = ids[s * B_SZ + b];
  atomicMax((int*)(cpy + (long)b * V_SZ + id), __float_as_int(val));
}

// 9) final log_softmax over V (one block per row)
__global__ void __launch_bounds__(1024) lsm_kernel(
    float* __restrict__ pred, const float* __restrict__ cpy,
    float* __restrict__ out) {
  __shared__ float red[32];
  int b = blockIdx.x, tid = threadIdx.x;
  float* prow = pred + (long)b * V_SZ;
  const float* crow = cpy + (long)b * V_SZ;
  float lmax = -3.4e38f;
  for (int i = tid; i < V_SZ; i += 1024) {
    float t = prow[i] + crow[i];
    prow[i] = t;
    lmax = fmaxf(lmax, t);
  }
  float m = blk_max(lmax, red, 32);
  float lsum = 0.f;
  for (int i = tid; i < V_SZ; i += 1024) lsum += expf(prow[i] - m);
  float s = blk_sum(lsum, red, 32);
  float lg = m + logf(s);
  for (int i = tid; i < V_SZ; i += 1024) out[(long)b * V_SZ + i] = prow[i] - lg;
}

// ---------------------------------------------------------------- launch

extern "C" void kernel_launch(void* const* d_in, const int* in_sizes, int n_in,
                              void* d_out, int out_size, void* d_ws, size_t ws_size,
                              hipStream_t stream) {
  (void)in_sizes; (void)n_in; (void)out_size; (void)ws_size;
  const int*   ids  = (const int*)  d_in[0];
  const float* x    = (const float*)d_in[1];
  const float* h    = (const float*)d_in[2];
  // d_in[3] = ans_enc_outs (unused by the reference)
  const float* text = (const float*)d_in[4];
  const float* Wih  = (const float*)d_in[5];
  const float* Whh  = (const float*)d_in[6];
  const float* bih  = (const float*)d_in[7];
  const float* bhh  = (const float*)d_in[8];
  const float* Wa   = (const float*)d_in[9];
  const float* ba   = (const float*)d_in[10];
  const float* vv   = (const float*)d_in[11];
  const float* Wc   = (const float*)d_in[12];
  const float* bc   = (const float*)d_in[13];
  const float* Wo   = (const float*)d_in[14];
  const float* bo   = (const float*)d_in[15];
  const float* Wp   = (const float*)d_in[16];
  const float* bp   = (const float*)d_in[17];
  float* out = (float*)d_out;

  float* ws    = (float*)d_ws;
  float* hnew  = ws;                                // 65536
  float* hproj = ws + 65536;                        // 65536
  float* attE  = ws + 131072;                       // 65536
  float* wc    = ws + 196608;                       // 65536
  float* cat   = ws + 262144;                       // 65536
  float* pgen  = ws + 327680;                       // 128
  float* pred  = ws + 327808;                       // 6.4M
  float* cpy   = ws + 327808 + (long)B_SZ * V_SZ;   // 6.4M
  unsigned* bbufWa  = (unsigned*)(ws + 327808 + 2L * B_SZ * V_SZ);  // 131072 dw
  unsigned* abufCat = bbufWa + 131072;                              // 32768 dw

  float* hnew_out = out + (long)B_SZ * V_SZ;

  pack_wae_kernel<<<512, 256, 0, stream>>>(Wa, bbufWa);
  gru_kernel     <<<32, 256, 0, stream>>>(x, h, Wih, Whh, bih, bhh, hnew, hnew_out);
  hproj_kernel   <<<32, 256, 0, stream>>>(hnew, Wa, ba, hproj);
  att_kernel     <<<512, 256, 0, stream>>>(text, bbufWa, hproj, vv, attE);
  ctx_kernel     <<<128, 256, 0, stream>>>(attE, text, wc);
  pgen_kernel    <<<128, 32, 0, stream>>>(wc, hnew, x, Wp, bp, pgen);
  concat_kernel  <<<32, 256, 0, stream>>>(wc, hnew, Wc, bc, cat);
  pack_cat_kernel<<<128, 256, 0, stream>>>(cat, abufCat);
  pred_kernel    <<<(V_SZ / 16 + 7) / 8, 256, 0, stream>>>(abufCat, Wo, bo, pgen, pred, cpy);
  scatter_kernel <<<(B_SZ * S_SZ) / 256, 256, 0, stream>>>(pgen, attE, ids, cpy);
  lsm_kernel     <<<B_SZ, 1024, 0, stream>>>(pred, cpy, out);
}
